// TPParallelLlamaAttention_53626961658205
// MI455X (gfx1250) — matrix operand order repaired and checked by reading, not compiled, not run on hardware
//
#include <hip/hip_runtime.h>

typedef __attribute__((ext_vector_type(16))) _Float16 v16h;
typedef __attribute__((ext_vector_type(4)))  _Float16 v4h;
typedef __attribute__((ext_vector_type(8)))  float    v8f;

#define S_LEN   4096
#define D_MODEL 2048
#define NHEADS  16
#define NKV     4
#define HD      128
#define GROUPS  (NHEADS / NKV)

#define WMMA_F32_F16(a, b, c) \
    __builtin_amdgcn_wmma_f32_16x16x32_f16(false, (a), false, (b), (short)0, (c), false, false)

static __device__ __forceinline__ v8f v8f_zero() {
    return (v8f){0.f, 0.f, 0.f, 0.f, 0.f, 0.f, 0.f, 0.f};
}

// ---------------------------------------------------------------------------
// GEMM: C[m, coloff+n] = sum_k A[m,k] * Bw[n,k]   (i.e. C = A * Bw^T)
// A: M x K f32 (lda == K), Bw: N x K f32 (ldb == K), C row stride = ldc.
// grid.x = M/128, grid.y = N/128, 256 threads (8 waves, 4x2 wave grid,
// each wave computes 32x64 via 2x4 WMMA 16x16 accumulators).
// Double-buffered LDS staging: one barrier per K-step, global loads of the
// next tile overlap the 8 WMMAs of the current tile.
// ---------------------------------------------------------------------------
__global__ __launch_bounds__(256)
void gemm_f16wmma(const float* __restrict__ A, const float* __restrict__ Bw,
                  float* __restrict__ C, int K, int ldc, int coloff)
{
    __shared__ _Float16 lA[2][128][40];   // 128 x 32 tiles (+8 pad), ping-pong
    __shared__ _Float16 lB[2][128][40];

    const int tid  = threadIdx.x;
    const int lane = tid & 31;
    const int wave = tid >> 5;
    const int wm   = (wave >> 1) * 32;
    const int wn   = (wave & 1) * 64;
    const int tm   = blockIdx.x * 128;
    const int tn   = blockIdx.y * 128;

    const int m16 = lane & 15;
    const int grp = lane >> 4;
    const int kgA = grp * 8;    // A-fragment K base (16-bit A 16x32 layout)
    const int kgB = grp * 16;   // unused; B fragment is gathered with kgA (same k order as A)

    v8f acc[2][4];
#pragma unroll
    for (int i = 0; i < 2; ++i)
#pragma unroll
        for (int j = 0; j < 4; ++j)
            acc[i][j] = v8f_zero();

    const int rowA = tid >> 3;        // 0..31
    const int cA   = (tid & 7) * 4;   // 0,4,...,28

    // ---- prologue: stage tile 0 into buffer 0 ----
#pragma unroll
    for (int it = 0; it < 4; ++it) {
        const int r = rowA + it * 32;
        float4 va = *(const float4*)&A[(size_t)(tm + r) * K + cA];
        v4h ha = { (_Float16)va.x, (_Float16)va.y, (_Float16)va.z, (_Float16)va.w };
        *(v4h*)&lA[0][r][cA] = ha;
        float4 vb = *(const float4*)&Bw[(size_t)(tn + r) * K + cA];
        v4h hb = { (_Float16)vb.x, (_Float16)vb.y, (_Float16)vb.z, (_Float16)vb.w };
        *(v4h*)&lB[0][r][cA] = hb;
    }
    __syncthreads();

    const int nk = K / 32;
    for (int kt = 0; kt < nk; ++kt) {
        const int buf  = kt & 1;
        const bool more = (kt + 1 < nk);

        // ---- issue global loads for the next tile (overlap with WMMA) ----
        float4 pa[4], pb[4];
        if (more) {
            const int k0 = (kt + 1) * 32;
#pragma unroll
            for (int it = 0; it < 4; ++it) {
                const int r = rowA + it * 32;
                pa[it] = *(const float4*)&A[(size_t)(tm + r) * K + k0 + cA];
                pb[it] = *(const float4*)&Bw[(size_t)(tn + r) * K + k0 + cA];
            }
        }

        // ---- gather fragments from current buffer ----
        v16h afr[2], bfr[4];
#pragma unroll
        for (int mf = 0; mf < 2; ++mf) {
            const _Float16* p = &lA[buf][wm + mf * 16 + m16][0];
#pragma unroll
            for (int i = 0; i < 8; ++i) afr[mf][i] = p[kgA + i];
#pragma unroll
            for (int i = 0; i < 8; ++i) afr[mf][8 + i] = p[16 + kgA + i];
        }
#pragma unroll
        for (int nf = 0; nf < 4; ++nf) {
            const _Float16* p = &lB[buf][wn + nf * 16 + m16][0];
#pragma unroll
            for (int i = 0; i < 16; ++i) bfr[nf][i] = p[(i < 8) ? (kgA + i) : (8 + kgA + i)];
        }

#pragma unroll
        for (int mf = 0; mf < 2; ++mf)
#pragma unroll
            for (int nf = 0; nf < 4; ++nf)
                acc[mf][nf] = WMMA_F32_F16(afr[mf], bfr[nf], acc[mf][nf]);

        // ---- convert + store next tile, single barrier per step ----
        if (more) {
            const int nb = buf ^ 1;
#pragma unroll
            for (int it = 0; it < 4; ++it) {
                const int r = rowA + it * 32;
                v4h ha = { (_Float16)pa[it].x, (_Float16)pa[it].y,
                           (_Float16)pa[it].z, (_Float16)pa[it].w };
                *(v4h*)&lA[nb][r][cA] = ha;
                v4h hb = { (_Float16)pb[it].x, (_Float16)pb[it].y,
                           (_Float16)pb[it].z, (_Float16)pb[it].w };
                *(v4h*)&lB[nb][r][cA] = hb;
            }
            __syncthreads();
        }
    }

    // ---- epilogue: C/D layout -> global (VGPR r holds M = r + 8*grp) ----
#pragma unroll
    for (int mf = 0; mf < 2; ++mf)
#pragma unroll
        for (int nf = 0; nf < 4; ++nf)
#pragma unroll
            for (int r = 0; r < 8; ++r) {
                const int m = tm + wm + mf * 16 + 8 * grp + r;
                const int n = tn + wn + nf * 16 + m16;
                C[(size_t)m * ldc + coloff + n] = acc[mf][nf][r];
            }
}

// ---------------------------------------------------------------------------
// RoPE + head split: qkv f32 (S x 3072: [Q 2048 | K 512 | V 512]) ->
//   q f16 [h][s][d], k f16 [kv][s][d], v f16 [kv][s][d]
// ---------------------------------------------------------------------------
__global__ __launch_bounds__(256)
void rope_split(const float* __restrict__ qkv, const float* __restrict__ cs,
                const float* __restrict__ sn, _Float16* __restrict__ qh,
                _Float16* __restrict__ kh, _Float16* __restrict__ vh)
{
    const int s = blockIdx.x;
    const int c = blockIdx.y * 256 + threadIdx.x;   // 0..3071
    const float* row = qkv + (size_t)s * 3072;

    if (c < 2048) {
        const int h = c >> 7, d = c & 127;
        const float x  = row[c];
        const float xr = (d < 64) ? -row[c + 64] : row[c - 64];
        const float y  = x * cs[s * HD + d] + xr * sn[s * HD + d];
        qh[((size_t)h * S_LEN + s) * HD + d] = (_Float16)y;
    } else if (c < 2560) {
        const int kc = c - 2048, kvh = kc >> 7, d = kc & 127;
        const float x  = row[c];
        const float xr = (d < 64) ? -row[c + 64] : row[c - 64];
        const float y  = x * cs[s * HD + d] + xr * sn[s * HD + d];
        kh[((size_t)kvh * S_LEN + s) * HD + d] = (_Float16)y;
    } else {
        const int vc = c - 2560, kvh = vc >> 7, d = vc & 127;
        vh[((size_t)kvh * S_LEN + s) * HD + d] = (_Float16)row[c];
    }
}

// ---------------------------------------------------------------------------
// Async staging of one 32-key x 128-hd f16 K/V block into LDS via the CDNA5
// async copy engine (GLOBAL_LOAD_ASYNC_TO_LDS_B128, tracked with ASYNCcnt).
// GVS addressing: saddr = 64-bit base (SGPR pair), vaddr = 32-bit byte offset.
// ---------------------------------------------------------------------------
static __device__ __forceinline__ void stage_kv_async(
    int tid, int kb, unsigned long long kbase, unsigned long long vbase,
    _Float16 (*sKb)[136], _Float16 (*sVb)[136])
{
#pragma unroll
    for (int it = 0; it < 2; ++it) {
        const int idx = tid + it * 256;             // 0..511
        const int row = idx >> 4;                   // key within block
        const int ch  = idx & 15;                   // chunk of 8 halves (16B)
        const unsigned goff = (unsigned)(((unsigned)(kb * 32 + row) * HD + ch * 8) * 2);
        const unsigned lk = (unsigned)(uintptr_t)&sKb[row][ch * 8];
        const unsigned lv = (unsigned)(uintptr_t)&sVb[row][ch * 8];
        asm volatile("global_load_async_to_lds_b128 %0, %1, %2"
                     :: "v"(lk), "v"(goff), "s"(kbase) : "memory");
        asm volatile("global_load_async_to_lds_b128 %0, %1, %2"
                     :: "v"(lv), "v"(goff), "s"(vbase) : "memory");
    }
}

static __device__ __forceinline__ void wait_async_keep(int keep) {
    if (keep) asm volatile("s_wait_asynccnt 4" ::: "memory");
    else      asm volatile("s_wait_asynccnt 0" ::: "memory");
}

// ---------------------------------------------------------------------------
// Flash attention (causal, GQA): grid (S/128 q-tiles, H heads), 256 threads.
// Each wave owns 16 query rows; online softmax; P->A-layout via LDS.
// K/V blocks double-buffered in LDS, filled by the async copy engine so the
// DMA for block kb+1 overlaps the WMMAs + softmax of block kb.
// ---------------------------------------------------------------------------
__global__ __launch_bounds__(256)
void flash_attn(const _Float16* __restrict__ qh, const _Float16* __restrict__ kh,
                const _Float16* __restrict__ vh, float* __restrict__ attn)
{
    __shared__ _Float16 sK[2][32][136];   // 32 keys x 128 hd (+pad), ping-pong
    __shared__ _Float16 sV[2][32][136];
    __shared__ _Float16 sP[8][16][40];    // per-wave P staging 16x32 (+pad)

    const int qt   = blockIdx.x;
    const int h    = blockIdx.y;
    const int kvh  = h / GROUPS;
    const int tid  = threadIdx.x;
    const int lane = tid & 31;
    const int wave = tid >> 5;
    const int m16  = lane & 15;
    const int grp  = lane >> 4;
    const int kgA  = grp * 8;
    const int kgB  = grp * 16;

    const float scale = 0.08838834764831845f;   // 1/sqrt(128)
    const int q0 = qt * 128 + wave * 16;        // first query row of this wave

    const unsigned long long kbase =
        (unsigned long long)(uintptr_t)(kh + (size_t)kvh * S_LEN * HD);
    const unsigned long long vbase =
        (unsigned long long)(uintptr_t)(vh + (size_t)kvh * S_LEN * HD);

    // ---- Q fragments: 16 rows x 128 hd = four 16x32 A fragments, pre-scaled
    v16h qf[4];
    {
        const _Float16* qp = qh + ((size_t)h * S_LEN + (q0 + m16)) * HD;
#pragma unroll
        for (int c = 0; c < 4; ++c) {
#pragma unroll
            for (int i = 0; i < 8; ++i)
                qf[c][i] = (_Float16)((float)qp[c * 32 + kgA + i] * scale);
#pragma unroll
            for (int i = 0; i < 8; ++i)
                qf[c][8 + i] = (_Float16)((float)qp[c * 32 + 16 + kgA + i] * scale);
        }
    }

    v8f oacc[8];
#pragma unroll
    for (int i = 0; i < 8; ++i) oacc[i] = v8f_zero();
    float mrow[8], lrow[8];
#pragma unroll
    for (int r = 0; r < 8; ++r) { mrow[r] = -__builtin_inff(); lrow[r] = 0.f; }

    const int nkb = (qt + 1) * 4;   // 32-key blocks up to the causal diagonal

    // prologue: kick off DMA for block 0 into buffer 0
    stage_kv_async(tid, 0, kbase, vbase, sK[0], sV[0]);

    for (int kb = 0; kb < nkb; ++kb) {
        const int buf  = kb & 1;
        const int more = (kb + 1 < nkb);
        // issue DMA for next block into the other buffer (reads of that
        // buffer finished before the tail barrier of the previous iteration)
        if (more)
            stage_kv_async(tid, kb + 1, kbase, vbase, sK[buf ^ 1], sV[buf ^ 1]);
        wait_async_keep(more);      // own block-kb ops done; next stays in flight
        __syncthreads();            // -> every wave's block-kb data is in LDS

        // ---- scores: two 16-key tiles, accumulate over HD in 4 chunks ----
        v8f s0 = v8f_zero(), s1 = v8f_zero();
#pragma unroll
        for (int c = 0; c < 4; ++c) {
            v16h b0, b1;
            const _Float16* pk0 = &sK[buf][m16][c * 32 + kgA];
            const _Float16* pk1 = &sK[buf][16 + m16][c * 32 + kgA];
#pragma unroll
            for (int i = 0; i < 16; ++i) { const int j = (i < 8) ? i : (i + 8); b0[i] = pk0[j]; b1[i] = pk1[j]; }
            s0 = WMMA_F32_F16(qf[c], b0, s0);
            s1 = WMMA_F32_F16(qf[c], b1, s1);
        }

        // ---- causal mask + online softmax (rows r+8*grp of this wave) ----
#pragma unroll
        for (int r = 0; r < 8; ++r) {
            const int qrow = q0 + 8 * grp + r;
            const int key0 = kb * 32 + m16;
            float a0 = (key0      <= qrow) ? s0[r] : -1.0e9f;
            float a1 = (key0 + 16 <= qrow) ? s1[r] : -1.0e9f;
            float lm = fmaxf(a0, a1);
#pragma unroll
            for (int off = 8; off >= 1; off >>= 1)
                lm = fmaxf(lm, __shfl_xor(lm, off, 32));
            const float nm = fmaxf(mrow[r], lm);
            const float sc = __expf(mrow[r] - nm);
            const float p0 = __expf(a0 - nm);
            const float p1 = __expf(a1 - nm);
            float rs = p0 + p1;
#pragma unroll
            for (int off = 8; off >= 1; off >>= 1)
                rs += __shfl_xor(rs, off, 32);
            mrow[r] = nm;
            lrow[r] = lrow[r] * sc + rs;
#pragma unroll
            for (int hf = 0; hf < 8; ++hf) oacc[hf][r] *= sc;
            const int prow = 8 * grp + r;
            sP[wave][prow][m16]      = (_Float16)p0;
            sP[wave][prow][m16 + 16] = (_Float16)p1;
        }
        __syncthreads();   // make sP visible in A-layout gather

        // ---- O += P x V ----
        v16h pf;
        {
            const _Float16* pp = &sP[wave][m16][0];
#pragma unroll
            for (int i = 0; i < 8; ++i) pf[i] = pp[kgA + i];
#pragma unroll
            for (int i = 0; i < 8; ++i) pf[8 + i] = pp[16 + kgA + i];
        }
#pragma unroll
        for (int hf = 0; hf < 8; ++hf) {
            v16h bv;
#pragma unroll
            for (int i = 0; i < 16; ++i) bv[i] = sV[buf][(i < 8) ? (kgA + i) : (8 + kgA + i)][hf * 16 + m16];
            oacc[hf] = WMMA_F32_F16(pf, bv, oacc[hf]);
        }
        __syncthreads();   // all sK/sV[buf^1? no: buf] reads done before next DMA
    }

    // ---- normalize + write: attn[s][h*HD + d] ----
#pragma unroll
    for (int hf = 0; hf < 8; ++hf)
#pragma unroll
        for (int r = 0; r < 8; ++r) {
            const int srow = q0 + 8 * grp + r;
            const int col  = h * HD + hf * 16 + m16;
            attn[(size_t)srow * D_MODEL + col] = oacc[hf][r] / lrow[r];
        }
}

// ---------------------------------------------------------------------------
extern "C" void kernel_launch(void* const* d_in, const int* in_sizes, int n_in,
                              void* d_out, int out_size, void* d_ws, size_t ws_size,
                              hipStream_t stream)
{
    (void)in_sizes; (void)n_in; (void)out_size; (void)ws_size;

    const float* hs = (const float*)d_in[0];
    const float* cs = (const float*)d_in[1];
    const float* sn = (const float*)d_in[2];
    // d_in[3] = attention_mask: pure causal -> applied analytically in-kernel
    const float* wq = (const float*)d_in[4];
    const float* wk = (const float*)d_in[5];
    const float* wv = (const float*)d_in[6];
    const float* wo = (const float*)d_in[7];

    char* ws = (char*)d_ws;
    float*    qkv = (float*)ws;                                   // 4096 x 3072 f32
    _Float16* qh  = (_Float16*)(ws + (size_t)S_LEN * 3072 * 4);   // 16 x 4096 x 128
    _Float16* kh  = qh + (size_t)NHEADS * S_LEN * HD;             //  4 x 4096 x 128
    _Float16* vh  = kh + (size_t)NKV * S_LEN * HD;                //  4 x 4096 x 128

    const dim3 blk(256);
    // QKV projections into one fused buffer (cols: Q[0,2048) K[2048,2560) V[2560,3072))
    gemm_f16wmma<<<dim3(32, 16), blk, 0, stream>>>(hs, wq, qkv, D_MODEL, 3072, 0);
    gemm_f16wmma<<<dim3(32,  4), blk, 0, stream>>>(hs, wk, qkv, D_MODEL, 3072, 2048);
    gemm_f16wmma<<<dim3(32,  4), blk, 0, stream>>>(hs, wv, qkv, D_MODEL, 3072, 2560);
    // RoPE + head split to f16 operands
    rope_split<<<dim3(S_LEN, 12), blk, 0, stream>>>(qkv, cs, sn, qh, kh, vh);
    // Causal GQA flash attention (reuses qkv buffer as f32 output, ld = 2048)
    float* attn = qkv;
    flash_attn<<<dim3(S_LEN / 128, NHEADS), blk, 0, stream>>>(qh, kh, vh, attn);
    // Output projection
    gemm_f16wmma<<<dim3(32, 16), blk, 0, stream>>>(attn, wo, (float*)d_out, D_MODEL, D_MODEL, 0);
}
